// LSTM_4518305595513
// MI455X (gfx1250) — matrix operand order repaired and checked by reading, not compiled, not run on hardware
//
#include <hip/hip_runtime.h>

typedef __attribute__((ext_vector_type(16))) _Float16 v16h;
typedef __attribute__((ext_vector_type(8)))  _Float16 v8h;
typedef __attribute__((ext_vector_type(8)))  float    v8f;

constexpr int BB = 1024;   // batch
constexpr int SS = 512;    // sequence length
constexpr int HH = 128;    // hidden
constexpr int CC = 10;     // classes
constexpr int BT = 16;     // batch tile per workgroup
constexpr int HSTR = 136;  // padded h row stride in elements (272B -> 4-bank rotate/row)

// ---- Fast activations -------------------------------------------------------
// CDNA5 has a native v_tanh_f32 TRANS op (co-executes with XDL WMMA).
// Probe for the builtin; fall back to v_exp_f32 + v_rcp_f32 (no slow exact-div).
#if __has_builtin(__builtin_amdgcn_tanhf)
__device__ __forceinline__ float fast_tanh(float x) {
    return __builtin_amdgcn_tanhf(x);
}
#elif __has_builtin(__builtin_amdgcn_tanh_f32)
__device__ __forceinline__ float fast_tanh(float x) {
    return __builtin_amdgcn_tanh_f32(x);
}
#else
__device__ __forceinline__ float fast_tanh(float x) {
    float e = __expf(-2.0f * x);                     // v_exp_f32
    return 1.0f - 2.0f * __builtin_amdgcn_rcpf(1.0f + e) * e; // v_rcp_f32
}
#endif

// sigmoid(x) = 0.5 * tanh(x/2) + 0.5  -> one TRANS + one FMA
__device__ __forceinline__ float fast_sigmoid(float x) {
    return __builtin_fmaf(0.5f, fast_tanh(0.5f * x), 0.5f);
}

__launch_bounds__(256, 1)
__global__ void lstm_persistent_wmma(
    const float* __restrict__ x,
    const float* __restrict__ Wgx, const float* __restrict__ Wgh,
    const float* __restrict__ Wix, const float* __restrict__ Wih,
    const float* __restrict__ Wfx, const float* __restrict__ Wfh,
    const float* __restrict__ Wox, const float* __restrict__ Woh,
    const float* __restrict__ Wph,
    const float* __restrict__ bg, const float* __restrict__ bi,
    const float* __restrict__ bfv, const float* __restrict__ bo,
    const float* __restrict__ bp,
    float* __restrict__ out)
{
    __shared__ __align__(16) _Float16 hbuf[2][BT * HSTR];  // ping-pong h state (f16)
    __shared__ float xt[BT * SS];                          // staged x tile

    const int tid  = threadIdx.x;
    const int lane = tid & 31;
    const int wave = tid >> 5;        // 0..7 -> hidden-column tile
    const int b0   = blockIdx.x * BT; // batch tile base
    const int hi   = (lane >= 16) ? 1 : 0;
    const int lm   = lane & 15;
    const int n    = wave * 16 + lm;  // hidden column owned by this lane

    // Stage x tile: 16 consecutive batch rows are contiguous in global memory.
    const float* xsrc = x + (size_t)b0 * SS;
    for (int i = tid; i < BT * SS; i += 256) xt[i] = xsrc[i];
    // Zero h ping-pong buffers (h0 = 0).
    for (int i = tid; i < BT * HSTR; i += 256) {
        hbuf[0][i] = (_Float16)0.0f;
        hbuf[1][i] = (_Float16)0.0f;
    }

    // Per-lane rank-1 input weights + biases for column n.
    const float wgx = Wgx[n], wix = Wix[n], wfx = Wfx[n], wox = Wox[n];
    const float bgl = bg[n],  bil = bi[n],  bfl = bfv[n], bol = bo[n];

    // Load recurrent weights into registers in WMMA B-fragment layout:
    // lane (lm,hi) holds column n; element e -> K = kt*32 + hi*16 + e.
    v16h wgf[4], wif[4], wff[4], wof[4];
    for (int kt = 0; kt < 4; ++kt) {
        const int kbase = kt * 32 + hi * 16;
        v16h g, ii, ff, oo;
        #pragma unroll
        for (int e = 0; e < 16; ++e) {
            const int k = kbase + e;
            g[e]  = (_Float16)Wgh[k * HH + n];
            ii[e] = (_Float16)Wih[k * HH + n];
            ff[e] = (_Float16)Wfh[k * HH + n];
            oo[e] = (_Float16)Woh[k * HH + n];
        }
        wgf[kt] = g; wif[kt] = ii; wff[kt] = ff; wof[kt] = oo;
    }

    // Cell state in registers, C/D accumulator layout: VGPR r -> row r + hi*8, col n.
    float cst[8];
    #pragma unroll
    for (int r = 0; r < 8; ++r) cst[r] = 0.0f;

    __syncthreads();

    #pragma unroll 1
    for (int t = 0; t < SS; ++t) {
        const _Float16* hr = hbuf[t & 1];
        _Float16*       hw = hbuf[(t + 1) & 1];

        v8f ag = {}, ai = {}, af = {}, ao = {};
        #pragma unroll
        for (int kt = 0; kt < 4; ++kt) {
            // A-fragment (contiguous k order, not ISA table order): lane row lm; low lanes take
            // K kt*32+0..15, high lanes kt*32+16..31 (matches B fragment).
            const _Float16* arow = hr + lm * HSTR + kt * 32 + hi * 16;
            v8h lo = *(const v8h*)(arow);
            v8h h8 = *(const v8h*)(arow + 8);
            v16h a = __builtin_shufflevector(lo, h8,
                     0, 1, 2, 3, 4, 5, 6, 7, 8, 9, 10, 11, 12, 13, 14, 15);
            ag = __builtin_amdgcn_wmma_f32_16x16x32_f16(false, a, false, wgf[kt], (short)0, ag, false, false);
            ai = __builtin_amdgcn_wmma_f32_16x16x32_f16(false, a, false, wif[kt], (short)0, ai, false, false);
            af = __builtin_amdgcn_wmma_f32_16x16x32_f16(false, a, false, wff[kt], (short)0, af, false, false);
            ao = __builtin_amdgcn_wmma_f32_16x16x32_f16(false, a, false, wof[kt], (short)0, ao, false, false);
        }

        // Elementwise gates + state update. Row m = r + hi*8, col n.
        #pragma unroll
        for (int r = 0; r < 8; ++r) {
            const int m = r + hi * 8;
            const float xv = xt[m * SS + t];
            const float gg = fast_tanh   (__builtin_fmaf(xv, wgx, ag[r] + bgl));
            const float iv = fast_sigmoid(__builtin_fmaf(xv, wix, ai[r] + bil));
            const float fv = fast_sigmoid(__builtin_fmaf(xv, wfx, af[r] + bfl));
            const float ov = fast_sigmoid(__builtin_fmaf(xv, wox, ao[r] + bol));
            const float cn = __builtin_fmaf(gg, iv, cst[r] * fv);
            cst[r] = cn;
            const float hn = fast_tanh(cn) * ov;
            hw[m * HSTR + n] = (_Float16)hn;
        }
        // Single barrier per step: ping-pong makes write-after-read safe.
        __syncthreads();
    }

    // Final h lives in hbuf[SS & 1] == hbuf[0]. Tiny 128x10 projection + softmax.
    if (tid < BT) {
        const _Float16* hrow = hbuf[SS & 1] + tid * HSTR;
        float p[CC];
        #pragma unroll
        for (int c = 0; c < CC; ++c) p[c] = bp[c];
        for (int k = 0; k < HH; ++k) {
            const float hv = (float)hrow[k];
            #pragma unroll
            for (int c = 0; c < CC; ++c) p[c] = __builtin_fmaf(hv, Wph[k * CC + c], p[c]);
        }
        float mx = p[0];
        #pragma unroll
        for (int c = 1; c < CC; ++c) mx = fmaxf(mx, p[c]);
        float s = 0.0f;
        #pragma unroll
        for (int c = 0; c < CC; ++c) { p[c] = __expf(p[c] - mx); s += p[c]; }
        const float inv = __builtin_amdgcn_rcpf(s);
        float* orow = out + (size_t)(b0 + tid) * CC;
        #pragma unroll
        for (int c = 0; c < CC; ++c) orow[c] = p[c] * inv;
    }
}

extern "C" void kernel_launch(void* const* d_in, const int* in_sizes, int n_in,
                              void* d_out, int out_size, void* d_ws, size_t ws_size,
                              hipStream_t stream) {
    const float* x   = (const float*)d_in[0];
    const float* Wgx = (const float*)d_in[1];
    const float* Wgh = (const float*)d_in[2];
    const float* Wix = (const float*)d_in[3];
    const float* Wih = (const float*)d_in[4];
    const float* Wfx = (const float*)d_in[5];
    const float* Wfh = (const float*)d_in[6];
    const float* Wox = (const float*)d_in[7];
    const float* Woh = (const float*)d_in[8];
    const float* Wph = (const float*)d_in[9];
    const float* bg  = (const float*)d_in[10];
    const float* bi  = (const float*)d_in[11];
    const float* bf  = (const float*)d_in[12];
    const float* bo  = (const float*)d_in[13];
    const float* bp  = (const float*)d_in[14];
    float* out = (float*)d_out;

    dim3 grid(BB / BT);   // 64 workgroups, each owns 16 batch rows persistently
    dim3 block(256);      // 8 waves: one hidden-column tile (16 cols x 4 gates) each
    lstm_persistent_wmma<<<grid, block, 0, stream>>>(
        x, Wgx, Wgh, Wix, Wih, Wfx, Wfh, Wox, Woh, Wph,
        bg, bi, bf, bo, bp, out);
}
